// PhysNetCore_46497315946665
// MI455X (gfx1250) — matrix operand order repaired and checked by reading, not compiled, not run on hardware
//
#include <hip/hip_runtime.h>

// ---------------------------------------------------------------------------
// PhysNetCore on MI455X (gfx1250).
// Memory-bound problem (~170MB HBM traffic -> ~7us floor at 23.3 TB/s).
// Gather table (pe, 12.8MB bf16) and scatter accumulator (v, 25.6MB f32)
// are L2-resident (192MB L2). All 64x64 matmuls run on
// v_wmma_f32_16x16x32_bf16 (bf16 A/B, f32 accum), issued back-to-back so
// WMMA->VALU hazard NOPs are hidden behind other WMMAs.  Softplus uses raw
// v_exp_f32/v_log_f32 (arg always in (1,2] -> no libm fixups needed).
// ---------------------------------------------------------------------------

typedef __bf16 bf16_t;
typedef __attribute__((ext_vector_type(16))) __bf16 v16bf;
typedef __attribute__((ext_vector_type(8)))  float  v8f;

#define F 64
#define ASTR 72   // bf16 LDS tile row stride (bank-conflict padding, 16B aligned)
#define VSTR 68   // f32  LDS tile row stride (bank-conflict padding)
#define WAVES 8   // waves per block (256 threads, wave32)

// bf16 weight blob layout in workspace (element offsets)
#define OFF_WG    0       // 64x16
#define OFF_WI    1024    // 64x64
#define OFF_WJ    5120
#define OFF_WV    9216
#define OFF_RIN1  13312   // 3 x 64x64
#define OFF_RIN2  25600   // 3 x 64x64
#define OFF_ROUT1 37888   // 2 x 64x64
#define OFF_ROUT2 46080   // 2 x 64x64
#define WTS_BYTES 131072  // rounded region size

#define LOG2E 1.4426950408889634f
#define LN2   0.6931471805599453f

// softplus(x) = max(x,0) + log(1 + exp(-|x|)).
// exp(-|x|) in (0,1] so the log argument is in (1,2]: numerically stable and
// safely in the raw-hardware-accuracy range of v_exp_f32/v_log_f32.
__device__ __forceinline__ float sp(float x) {
  float t = __builtin_amdgcn_exp2f(-LOG2E * fabsf(x));
  return fmaxf(x, 0.0f) + LN2 * __builtin_amdgcn_logf(1.0f + t);
}

__device__ __forceinline__ v8f wmma_bf16(v16bf a, v16bf b, v8f c) {
  return __builtin_amdgcn_wmma_f32_16x16x32_bf16(
      /*neg_a=*/false, a, /*neg_b=*/false, b, /*c_mod=*/(short)0, c,
      /*reuse_a=*/false, /*reuse_b=*/false);
}

// A fragment (16x32 bf16, MxK) from LDS tile (row stride ASTR).
// ISA 7.12.2: lanes 0-15 hold M=lane, K {kt*32+0..7, kt*32+16..23};
//             lanes 16-31 hold M=lane-16, K {kt*32+8..15, kt*32+24..31}.
__device__ __forceinline__ v16bf ldsA(const bf16_t* tile, int lane, int kt) {
  const bf16_t* p = tile + (lane & 15) * ASTR + kt * 32 + ((lane & 16) ? 8 : 0);
  union { v16bf v; uint4 q[2]; } u;
  u.q[0] = *(const uint4*)(p);
  u.q[1] = *(const uint4*)(p + 16);
  return u.v;
}

// B fragment (32x16 bf16, KxN) for OUT = X @ W^T, W row-major [64][64] bf16.
// B[k][n] = W[n][k]. lanes 0-15: N=lane, K {kt*32+0..7, kt*32+16..23};
// lanes 16-31: N=lane-16, K {kt*32+8..15, kt*32+24..31}.
__device__ __forceinline__ v16bf glbB(const bf16_t* W, int lane, int kt, int nt) {
  const bf16_t* p = W + (size_t)(nt * 16 + (lane & 15)) * F + kt * 32 + ((lane & 16) ? 8 : 0);
  union { v16bf v; uint4 q[2]; } u;
  u.q[0] = *(const uint4*)(p);
  u.q[1] = *(const uint4*)(p + 16);
  return u.v;
}

// OUT[16,64] = A(tile) @ W^T.  All 8 WMMAs are issued before any epilogue
// reads an accumulator, hiding the WMMA->VALU hazard window with real work.
// D layout: lane holds col N = nt*16 + (lane&15), rows M = r + (lane&16 ? 8:0).
template <class Epi>
__device__ __forceinline__ void mm64(const bf16_t* tile, const bf16_t* W, int lane, Epi epi) {
  __builtin_amdgcn_wave_barrier();
  v16bf a0 = ldsA(tile, lane, 0);
  v16bf a1 = ldsA(tile, lane, 1);
  v8f acc[4];
#pragma unroll
  for (int nt = 0; nt < 4; ++nt) {
    v8f c = {};
    c = wmma_bf16(a0, glbB(W, lane, 0, nt), c);
    acc[nt] = wmma_bf16(a1, glbB(W, lane, 1, nt), c);
  }
#pragma unroll
  for (int nt = 0; nt < 4; ++nt) epi(nt, acc[nt]);
  __builtin_amdgcn_wave_barrier();
}

// PhysNet residual block, fused: on entry ab = bf16(sp(vb)); on exit
// vb = vb + sp(sp(vb)@W1^T+b1)@W2^T + b2 and ab = bf16(sp(vb_new)).
__device__ __forceinline__ void residual(float* vb, bf16_t* ab,
                                         const bf16_t* W1, const float* b1,
                                         const bf16_t* W2, const float* b2, int lane) {
  mm64(ab, W1, lane, [&](int nt, v8f acc) {
    int col = nt * 16 + (lane & 15);
    float bb = b1[col];
    int rb = (lane & 16) ? 8 : 0;
#pragma unroll
    for (int r = 0; r < 8; ++r)
      ab[(rb + r) * ASTR + col] = (bf16_t)sp(acc[r] + bb);
  });
  mm64(ab, W2, lane, [&](int nt, v8f acc) {
    int col = nt * 16 + (lane & 15);
    float bb = b2[col];
    int rb = (lane & 16) ? 8 : 0;
#pragma unroll
    for (int r = 0; r < 8; ++r) {
      int row = rb + r;
      float nv = vb[row * VSTR + col] + acc[r] + bb;
      vb[row * VSTR + col] = nv;
      ab[row * ASTR + col] = (bf16_t)sp(nv);
    }
  });
}

// ---------------------------------------------------------------------------
// K0: convert all weight matrices to bf16 blob in workspace (tiny, one-shot).
// ---------------------------------------------------------------------------
__global__ void k_convw(const float* wg, const float* wi, const float* wj, const float* wv,
                        const float* rw1, const float* rw2, const float* ow1, const float* ow2,
                        bf16_t* dst) {
  int t = blockIdx.x * blockDim.x + threadIdx.x;
  int s = gridDim.x * blockDim.x;
  for (int i = t; i < 1024; i += s) dst[OFF_WG + i] = (bf16_t)wg[i];
  for (int i = t; i < 4096; i += s) {
    dst[OFF_WI + i] = (bf16_t)wi[i];
    dst[OFF_WJ + i] = (bf16_t)wj[i];
    dst[OFF_WV + i] = (bf16_t)wv[i];
  }
  for (int i = t; i < 3 * 4096; i += s) {
    dst[OFF_RIN1 + i] = (bf16_t)rw1[i];
    dst[OFF_RIN2 + i] = (bf16_t)rw2[i];
  }
  for (int i = t; i < 2 * 4096; i += s) {
    dst[OFF_ROUT1 + i] = (bf16_t)ow1[i];
    dst[OFF_ROUT2 + i] = (bf16_t)ow2[i];
  }
}

// ---------------------------------------------------------------------------
// K1: pe = softplus(emb) (store f32 + bf16 copies); v = softplus(pe@W_i^T+b_i)
// ---------------------------------------------------------------------------
__global__ void __launch_bounds__(256) k_atoms_pre(const float* __restrict__ emb,
                                                   const bf16_t* __restrict__ wts,
                                                   const float* __restrict__ b_i,
                                                   float* __restrict__ ws_pe,
                                                   bf16_t* __restrict__ ws_peb,
                                                   float* __restrict__ ws_v, int N) {
  __shared__ __align__(16) bf16_t abuf_s[WAVES][16 * ASTR];
  int wave = threadIdx.x >> 5, lane = threadIdx.x & 31;
  int base = (blockIdx.x * WAVES + wave) * 16;
  if (base >= N) return;
  const bool full = (base + 16 <= N);  // wave-uniform fast path
  bf16_t* ab = abuf_s[wave];

  if (full) {
    const float* src = emb + (size_t)base * F;
    float* pdst = ws_pe + (size_t)base * F;
    bf16_t* bdst = ws_peb + (size_t)base * F;
#pragma unroll
    for (int i = lane; i < 16 * F; i += 32) {
      int m = i >> 6, c = i & 63;
      float pe = sp(src[i]);
      ab[m * ASTR + c] = (bf16_t)pe;
      pdst[i] = pe;
      bdst[i] = (bf16_t)pe;
    }
  } else {
    for (int i = lane; i < 16 * F; i += 32) {
      int m = i >> 6, c = i & 63;
      int atom = base + m;
      int ca = atom < N ? atom : N - 1;
      float pe = sp(emb[(size_t)ca * F + c]);
      ab[m * ASTR + c] = (bf16_t)pe;
      if (atom < N) {
        ws_pe[(size_t)atom * F + c] = pe;
        ws_peb[(size_t)atom * F + c] = (bf16_t)pe;
      }
    }
  }
  mm64(ab, wts + OFF_WI, lane, [&](int nt, v8f acc) {
    int col = nt * 16 + (lane & 15);
    float bb = b_i[col];
    int rb = (lane & 16) ? 8 : 0;
    float* dst = ws_v + (size_t)(base + rb) * F + col;  // constant r*F offsets
    if (full) {
#pragma unroll
      for (int r = 0; r < 8; ++r) dst[r * F] = sp(acc[r] + bb);
    } else {
#pragma unroll
      for (int r = 0; r < 8; ++r)
        if (base + rb + r < N) dst[r * F] = sp(acc[r] + bb);
    }
  });
}

// ---------------------------------------------------------------------------
// K2: pair stream.  msg = softplus(pe[idx_j]@W_j^T + b_j) * (f_ij@W_g^T),
// scatter-add into v[idx_i].  16 pairs per wave, grid-stride with prefetch.
// B fragments (weights) are loop-invariant and held in VGPRs; all 12 WMMAs
// per tile issue before any accumulator is consumed.
// ---------------------------------------------------------------------------
__global__ void __launch_bounds__(256) k_pairs(const float* __restrict__ f_ij,
                                               const int* __restrict__ pidx,
                                               const bf16_t* __restrict__ wts,
                                               const bf16_t* __restrict__ ws_peb,
                                               const float* __restrict__ b_j,
                                               float* __restrict__ ws_v, int P) {
  __shared__ int idx_s[WAVES][16];
  int wave = threadIdx.x >> 5, lane = threadIdx.x & 31;
  int m16 = lane & 15;
  int rb = (lane & 16) ? 8 : 0;
  long ntiles = ((long)P + 15) >> 4;
  long tstep = (long)gridDim.x * WAVES;

  // --- hoisted, loop-invariant weight fragments + biases ---
  v16bf bj0[4], bj1[4], bg[4];
  float bjv[4];
#pragma unroll
  for (int nt = 0; nt < 4; ++nt) {
    bj0[nt] = glbB(wts + OFF_WJ, lane, 0, nt);
    bj1[nt] = glbB(wts + OFF_WJ, lane, 1, nt);
    // radial gate B: K rows 16..31 are zero (elements 8..15 of every lane hold zeros)
    union { v16bf v; uint4 q[2]; } u;
    {
      const bf16_t* p = wts + OFF_WG + (size_t)(nt * 16 + m16) * 16 + ((lane & 16) ? 8 : 0);
      u.q[0] = *(const uint4*)(p);
      u.q[1] = make_uint4(0, 0, 0, 0);
    }
    bg[nt] = u.v;
    bjv[nt] = b_j[nt * 16 + m16];
  }

  for (long tile = (long)blockIdx.x * WAVES + wave; tile < ntiles; tile += tstep) {
    long p0 = tile * 16;
    const bool fullt = (p0 + 16 <= (long)P);  // wave-uniform
    long pm = p0 + m16;
    if (pm >= P) pm = P - 1;

    // prefetch next tile's radial features + indices (hides HBM latency)
    long pnext = pm + tstep * 16;
    if (pnext < P) {
      __builtin_prefetch(f_ij + pnext * 16, 0, 0);
      __builtin_prefetch(pidx + pnext, 0, 0);
    }

    int aj = pidx[(size_t)P + pm];                // idx_j (gather row)
    if (lane < 16) idx_s[wave][lane] = pidx[pm];  // idx_i (scatter row)
    __builtin_amdgcn_wave_barrier();

    // A fragments for gathered pe rows (bf16, L2-resident table)
    const bf16_t* prow = ws_peb + (size_t)aj * F + ((lane & 16) ? 8 : 0);
    v16bf apj0, apj1;
    {
      union { v16bf v; uint4 q[2]; } u;
      u.q[0] = *(const uint4*)(prow);      u.q[1] = *(const uint4*)(prow + 16); apj0 = u.v;
      u.q[0] = *(const uint4*)(prow + 32); u.q[1] = *(const uint4*)(prow + 48); apj1 = u.v;
    }
    // A fragment for f_ij (K=16 zero-padded to 32), two 16B vector loads
    v16bf af;
    {
      const float4* frow = (const float4*)(f_ij + pm * 16 + ((lane & 16) ? 8 : 0));
      float4 f0 = frow[0], f1 = frow[1];
      af[0] = (bf16_t)f0.x; af[1] = (bf16_t)f0.y; af[2] = (bf16_t)f0.z; af[3] = (bf16_t)f0.w;
      af[4] = (bf16_t)f1.x; af[5] = (bf16_t)f1.y; af[6] = (bf16_t)f1.z; af[7] = (bf16_t)f1.w;
#pragma unroll
      for (int j = 8; j < 16; ++j) af[j] = (bf16_t)0.0f;
    }
    // scatter rows for this lane's half-tile (8 LDS reads per tile)
    int ai[8];
#pragma unroll
    for (int r = 0; r < 8; ++r) ai[r] = idx_s[wave][rb + r];

    // --- all 12 WMMAs back-to-back ---
    v8f macc[4], gacc[4];
#pragma unroll
    for (int nt = 0; nt < 4; ++nt) {
      v8f c = {};
      c = wmma_bf16(apj0, bj0[nt], c);
      macc[nt] = wmma_bf16(apj1, bj1[nt], c);
      v8f g = {};
      gacc[nt] = wmma_bf16(af, bg[nt], g);
    }

    // --- epilogue: softplus * gate, fp32 scatter atomics (L2-resident) ---
#pragma unroll
    for (int nt = 0; nt < 4; ++nt) {
      unsigned col = (unsigned)(nt * 16 + m16);
      float bb = bjv[nt];
      if (fullt) {
#pragma unroll
        for (int r = 0; r < 8; ++r) {
          float val = sp(macc[nt][r] + bb) * gacc[nt][r];
          unsigned off = (unsigned)ai[r] * 64u + col;  // 32-bit voffset + saddr
          __hip_atomic_fetch_add(ws_v + off, val,
                                 __ATOMIC_RELAXED, __HIP_MEMORY_SCOPE_AGENT);
        }
      } else {
#pragma unroll
        for (int r = 0; r < 8; ++r) {
          if (p0 + rb + r < P) {
            float val = sp(macc[nt][r] + bb) * gacc[nt][r];
            unsigned off = (unsigned)ai[r] * 64u + col;
            __hip_atomic_fetch_add(ws_v + off, val,
                                   __ATOMIC_RELAXED, __HIP_MEMORY_SCOPE_AGENT);
          }
        }
      }
    }
    __builtin_amdgcn_wave_barrier();
  }
}

// ---------------------------------------------------------------------------
// K3: residual chains + outputs.  11 WMMA matmuls per 16-atom tile, fully
// fused through per-wave LDS tiles.
// ---------------------------------------------------------------------------
__global__ void __launch_bounds__(256) k_atoms_post(
    const float* __restrict__ ws_v, const float* __restrict__ ws_pe,
    const bf16_t* __restrict__ wts,
    const float* __restrict__ rin_b1, const float* __restrict__ rin_b2,
    const float* __restrict__ rout_b1, const float* __restrict__ rout_b2,
    const float* __restrict__ b_v, const float* __restrict__ gate,
    const float* __restrict__ W_out, const float* __restrict__ b_out,
    float* __restrict__ out_pred, float* __restrict__ out_ue, int N) {
  __shared__ __align__(16) float  vbuf_s[WAVES][16 * VSTR];
  __shared__ __align__(16) bf16_t abuf_s[WAVES][16 * ASTR];
  int wave = threadIdx.x >> 5, lane = threadIdx.x & 31;
  int base = (blockIdx.x * WAVES + wave) * 16;
  if (base >= N) return;
  const bool full = (base + 16 <= N);
  float*  vb = vbuf_s[wave];
  bf16_t* ab = abuf_s[wave];

  if (full) {
    const float* src = ws_v + (size_t)base * F;
#pragma unroll
    for (int i = lane; i < 16 * F; i += 32) {
      int m = i >> 6, c = i & 63;
      float v = src[i];
      vb[m * VSTR + c] = v;
      ab[m * ASTR + c] = (bf16_t)sp(v);
    }
  } else {
    for (int i = lane; i < 16 * F; i += 32) {
      int m = i >> 6, c = i & 63;
      int atom = base + m;
      int ca = atom < N ? atom : N - 1;
      float v = ws_v[(size_t)ca * F + c];
      vb[m * VSTR + c] = v;
      ab[m * ASTR + c] = (bf16_t)sp(v);
    }
  }

  for (int r = 0; r < 3; ++r)  // interaction residual stack
    residual(vb, ab, wts + OFF_RIN1 + r * 4096, rin_b1 + r * F,
             wts + OFF_RIN2 + r * 4096, rin_b2 + r * F, lane);

  // updated_embedding = gate*pe + softplus(v)@W_v^T + b_v   (ab == bf16(sp(v)))
  mm64(ab, wts + OFF_WV, lane, [&](int nt, v8f acc) {
    int col = nt * 16 + (lane & 15);
    float bb = b_v[col], gg = gate[col];
    int rbl = (lane & 16) ? 8 : 0;
    const float* pesrc = ws_pe + (size_t)(base + rbl) * F + col;
    float* uedst = out_ue + (size_t)(base + rbl) * F + col;
    if (full) {
#pragma unroll
      for (int r = 0; r < 8; ++r) {
        int row = rbl + r;
        float ue = gg * pesrc[r * F] + acc[r] + bb;
        vb[row * VSTR + col] = ue;
        ab[row * ASTR + col] = (bf16_t)sp(ue);
        uedst[r * F] = ue;
      }
    } else {
#pragma unroll
      for (int r = 0; r < 8; ++r) {
        int row = rbl + r;
        int atom = base + row;
        int ca = atom < N ? atom : N - 1;
        float pe = ws_pe[(size_t)ca * F + col];
        float ue = gg * pe + acc[r] + bb;
        vb[row * VSTR + col] = ue;
        ab[row * ASTR + col] = (bf16_t)sp(ue);
        if (atom < N) out_ue[(size_t)atom * F + col] = ue;
      }
    }
  });

  for (int r = 0; r < 2; ++r)  // output residual stack
    residual(vb, ab, wts + OFF_ROUT1 + r * 4096, rout_b1 + r * F,
             wts + OFF_ROUT2 + r * 4096, rout_b2 + r * F, lane);
  __builtin_amdgcn_wave_barrier();

  // prediction head [64 -> 2]: lane -> (row m = lane&15, out a = lane>>4)
  {
    int m = lane & 15, a = lane >> 4;
    float s = b_out[a];
#pragma unroll 8
    for (int k = 0; k < F; ++k) s += vb[m * VSTR + k] * W_out[a * F + k];
    int atom = base + m;
    if (atom < N) out_pred[(size_t)atom * 2 + a] = s;
  }
}

// ---------------------------------------------------------------------------
extern "C" void kernel_launch(void* const* d_in, const int* in_sizes, int n_in,
                              void* d_out, int out_size, void* d_ws, size_t ws_size,
                              hipStream_t stream) {
  (void)n_in; (void)out_size; (void)ws_size;
  const int N = in_sizes[0] / F;   // 100000
  const int P = in_sizes[1] / 16;  // 1600000

  const float* emb     = (const float*)d_in[0];
  const float* f_ij    = (const float*)d_in[1];
  const int*   pidx    = (const int*)  d_in[2];
  const float* W_g     = (const float*)d_in[3];
  const float* W_i     = (const float*)d_in[4];
  const float* b_i     = (const float*)d_in[5];
  const float* W_j     = (const float*)d_in[6];
  const float* b_j     = (const float*)d_in[7];
  const float* W_v     = (const float*)d_in[8];
  const float* b_v     = (const float*)d_in[9];
  const float* gate    = (const float*)d_in[10];
  const float* rin_W1  = (const float*)d_in[11];
  const float* rin_b1  = (const float*)d_in[12];
  const float* rin_W2  = (const float*)d_in[13];
  const float* rin_b2  = (const float*)d_in[14];
  const float* rout_W1 = (const float*)d_in[15];
  const float* rout_b1 = (const float*)d_in[16];
  const float* rout_W2 = (const float*)d_in[17];
  const float* rout_b2 = (const float*)d_in[18];
  const float* W_out   = (const float*)d_in[19];
  const float* b_out   = (const float*)d_in[20];

  // workspace layout
  bf16_t* wts    = (bf16_t*)d_ws;
  float*  ws_pe  = (float*)((char*)d_ws + WTS_BYTES);
  float*  ws_v   = ws_pe + (size_t)N * F;
  bf16_t* ws_peb = (bf16_t*)(ws_v + (size_t)N * F);

  float* out_pred = (float*)d_out;             // [N,2]
  float* out_ue   = out_pred + (size_t)N * 2;  // [N,64]

  k_convw<<<64, 256, 0, stream>>>(W_g, W_i, W_j, W_v, rin_W1, rin_W2, rout_W1, rout_W2, wts);

  int atiles  = (N + 15) / 16;
  int ablocks = (atiles + WAVES - 1) / WAVES;
  k_atoms_pre<<<ablocks, 256, 0, stream>>>(emb, wts, b_i, ws_pe, ws_peb, ws_v, N);

  long ptiles  = ((long)P + 15) / 16;
  int  pblocks = (int)((ptiles + WAVES - 1) / WAVES);
  if (pblocks > 8192) pblocks = 8192;  // grid-stride in kernel
  k_pairs<<<pblocks, 256, 0, stream>>>(f_ij, pidx, wts, ws_peb, b_j, ws_v, P);

  k_atoms_post<<<ablocks, 256, 0, stream>>>(ws_v, ws_pe, wts, rin_b1, rin_b2, rout_b1, rout_b2,
                                            b_v, gate, W_out, b_out, out_pred, out_ue, N);
}
